// net_prot_gat_6545530159683
// MI455X (gfx1250) — compile-verified, matrix operand order checked by reading, not run on hardware
//
#include <hip/hip_runtime.h>

#define BATCH 64
#define NN    512
#define DD    256
#define LL    4
#define MTOT  (BATCH * NN)

typedef __attribute__((ext_vector_type(16))) __bf16         v16bf;
typedef __attribute__((ext_vector_type(8)))  float          v8f;
typedef __attribute__((ext_vector_type(4)))  float          v4f;
typedef __attribute__((ext_vector_type(8)))  unsigned short us8;
typedef __attribute__((ext_vector_type(4)))  unsigned short us4;
typedef __attribute__((ext_vector_type(4)))  unsigned int   v4u;
typedef __attribute__((ext_vector_type(8)))  int            v8i;
typedef __attribute__((ext_vector_type(4)))  int            v4i;

union Frag {
  v16bf bf;
  us8   h[2];
  unsigned short s[16];
};

__device__ __forceinline__ unsigned short f2bf(float f) {
  unsigned int u = __builtin_bit_cast(unsigned int, f);
  u += 0x7FFFu + ((u >> 16) & 1u);   // round-to-nearest-even
  return (unsigned short)(u >> 16);
}

// low 32 bits of the generic (flat) address of a __shared__ object = LDS byte offset
__device__ __forceinline__ unsigned lds_off(const void* p) {
  return (unsigned)(unsigned long long)p;
}

// ---------------------------------------------------------------------------
// TDM: async DMA of a 2D bf16 tile (tile_w x tile_h elems) from a row-major
// global tensor (row stride = stride_e elems) into LDS, with optional LDS row
// padding (pad_amount DWORDs every (1<<pad_interval) 8-byte units).
// Descriptor layout per CDNA5 ISA 8.3/8.4 (D# groups 0/1; 2D -> groups 2,3 = 0).
// This toolchain's builtin takes 6 args (extra trailing i32x8 group, zeroed).
// ---------------------------------------------------------------------------
__device__ __forceinline__ void tdm_load_2d(const void* gptr, unsigned lds_byte_off,
                                            unsigned tile_w, unsigned tile_h,
                                            unsigned tensor_w, unsigned tensor_h,
                                            unsigned stride_e,
                                            unsigned pad_interval, unsigned pad_amount)
{
  unsigned long long ga = (unsigned long long)gptr;
  union { unsigned int w[4]; v4u v; } g0;
  g0.w[0] = 1u;                                        // count=1 (valid user D#)
  g0.w[1] = lds_byte_off;                              // lds_addr (bytes)
  g0.w[2] = (unsigned)(ga & 0xFFFFFFFFu);              // global_addr[31:0]
  g0.w[3] = (unsigned)((ga >> 32) & 0x1FFFFFFu)        // global_addr[56:32]
          | (2u << 30);                                // type = 2 ("image")
  union { int w[8]; v8i v; } g1;
  unsigned pe = pad_amount ? 1u : 0u;
  g1.w[0] = (int)((1u << 16)                           // data_size = 1 -> 2 bytes
          | (pe << 20)                                 // pad_enable
          | ((pad_interval & 7u) << 22)                // pad_interval
          | ((pad_amount & 127u) << 25));              // pad_amount (DWORDs, -1 enc)
  g1.w[1] = (int)((tensor_w & 0xFFFFu) << 16);         // tensor_dim0[15:0] @ bits63:48
  g1.w[2] = (int)(((tensor_w >> 16) & 0xFFFFu)         // tensor_dim0[31:16]
          | ((tensor_h & 0xFFFFu) << 16));             // tensor_dim1[15:0]
  g1.w[3] = (int)(((tensor_h >> 16) & 0xFFFFu)         // tensor_dim1[31:16]
          | ((tile_w & 0xFFFFu) << 16));               // tile_dim0
  g1.w[4] = (int)(tile_h & 0xFFFFu);                   // tile_dim1 (tile_dim2=0)
  g1.w[5] = (int)stride_e;                             // tensor_dim0_stride[31:0]
  g1.w[6] = 0;                                         // stride[47:32], dim1_stride lo
  g1.w[7] = 0;
  v4i z4 = {0, 0, 0, 0};
  v8i z8 = {0, 0, 0, 0, 0, 0, 0, 0};
  __builtin_amdgcn_tensor_load_to_lds(g0.v, g1.v, z4, z4, z8, 0);
}

// ---------------------------------------------------------------------------
// One-shot f32 -> bf16 conversion (n multiple of 1024; 4 elems/thread)
// ---------------------------------------------------------------------------
__global__ __launch_bounds__(256)
void f32_to_bf16(const float* __restrict__ src, unsigned short* __restrict__ dst)
{
  int i = (blockIdx.x * 256 + threadIdx.x) * 4;
  v4f v = *(const v4f*)(src + i);
  us4 o;
  o.x = f2bf(v.x); o.y = f2bf(v.y); o.z = f2bf(v.z); o.w = f2bf(v.w);
  *(us4*)(dst + i) = o;
}

// ---------------------------------------------------------------------------
// Y = act(A[M,256]bf16 @ W[256,256]bf16^T + bias) (+res f32)
// Dual store: Yf (f32, optional) and Yb (bf16, optional).
// Block = 256 threads (8 waves), tile 64 rows x 128 cols, K step 32.
// Tiles staged by the Tensor Data Mover, double-buffered: wave 0 issues the
// next tile's DMA while all waves run WMMAs on the current one.
// ---------------------------------------------------------------------------
__global__ __launch_bounds__(256)
void gemm_bf16(const unsigned short* __restrict__ A, const unsigned short* __restrict__ W,
               const float* __restrict__ bias, const float* __restrict__ res,
               float* __restrict__ Yf, unsigned short* __restrict__ Yb, int act)
{
  const int K = DD, NO = DD;
  // padded rows: 32 elems + 4-DWORD pad -> 40 halfwords (TDM pad 16B per 64B)
  __shared__ __align__(16) unsigned short lx[2][64 * 40];
  __shared__ __align__(16) unsigned short lw[2][128 * 40];

  const int tid  = threadIdx.x;
  const int lane = tid & 31, wid = tid >> 5;
  const int wm   = wid & 3,  wn  = wid >> 2;              // 4x2 wave grid
  const int row0 = blockIdx.x * 64;
  const int col0 = blockIdx.y * 128;
  const int m    = lane & 15, half = lane >> 4;

  if (wid == 0) {   // prologue: DMA tile 0
    tdm_load_2d(A + (size_t)row0 * K, lds_off(&lx[0][0]), 32, 64,  K, MTOT, K, 3, 3);
    tdm_load_2d(W + (size_t)col0 * K, lds_off(&lw[0][0]), 32, 128, K, DD,   K, 3, 3);
  }

  v8f acc[4] = {};
  for (int i = 0; i < 8; ++i) {
    if (wid == 0) __builtin_amdgcn_s_wait_tensorcnt(0);   // tile i landed in LDS
    __syncthreads();
    if (wid == 0 && i < 7) {                              // DMA tile i+1 into other buf
      int kb = (i + 1) * 32;
      tdm_load_2d(A + (size_t)row0 * K + kb, lds_off(&lx[(i + 1) & 1][0]),
                  32, 64,  K, MTOT, K, 3, 3);
      tdm_load_2d(W + (size_t)col0 * K + kb, lds_off(&lw[(i + 1) & 1][0]),
                  32, 128, K, DD,   K, 3, 3);
    }
    const unsigned short* bx = lx[i & 1];
    const unsigned short* bw = lw[i & 1];

    Frag a;  // A: 16x32 bf16, lane=row, K per ISA layout
    a.h[0] = *(const us8*)&bx[(wm * 16 + m) * 40 + half * 8];
    a.h[1] = *(const us8*)&bx[(wm * 16 + m) * 40 + 16 + half * 8];
    #pragma unroll
    for (int t = 0; t < 4; ++t) {
      Frag b;  // B[k][n] = W[n][k] -> same access pattern on lw
      int cr = wn * 64 + t * 16 + m;
      b.h[0] = *(const us8*)&bw[cr * 40 + half * 8];
      b.h[1] = *(const us8*)&bw[cr * 40 + 16 + half * 8];
      acc[t] = __builtin_amdgcn_wmma_f32_16x16x32_bf16(
                   false, a.bf, false, b.bf, (short)0, acc[t], false, false);
    }
    __syncthreads();
  }

  #pragma unroll
  for (int t = 0; t < 4; ++t) {
    int cg = col0 + wn * 64 + t * 16 + m;
    float bv = bias[cg];
    #pragma unroll
    for (int r = 0; r < 8; ++r) {
      int rg = row0 + wm * 16 + r + 8 * half;
      float v = acc[t][r] + bv;
      if (act) v = v > 0.f ? v : 0.f;
      if (res) v += res[(size_t)rg * NO + cg];
      if (Yf)  Yf[(size_t)rg * NO + cg] = v;
      if (Yb)  Yb[(size_t)rg * NO + cg] = f2bf(v);
    }
  }
}

// ---------------------------------------------------------------------------
// Fused attention + aggregation (single pass — sigmoid attn needs only rowsum):
//   attn = (sigmoid(q x^T) + 1e-5 I) * adj(diag=1);  out = rownorm(attn) @ x
// q, x in bf16 (TDM-staged to LDS); out stored bf16.
// Block = 256 threads (8 waves) -> 32 rows of one batch; j streamed by 64.
// ---------------------------------------------------------------------------
#define QS 264   // q_lds row stride: 256 elems + 4-DWORD TDM pad
#define XS 264   // x_lds row stride
#define SS 72    // s_lds row stride

__global__ __launch_bounds__(256)
void gat_attn_agg(const unsigned short* __restrict__ q, const unsigned short* __restrict__ x,
                  const float* __restrict__ adj, unsigned short* __restrict__ out)
{
  __shared__ __align__(16) unsigned short q_lds[32 * QS];  // 32x256 bf16
  __shared__ __align__(16) unsigned short x_lds[64 * XS];  // 64x256 bf16
  __shared__ __align__(16) unsigned short s_lds[32 * SS];  // 32x64  bf16
  __shared__ float rowsum[32];

  const int tid  = threadIdx.x, lane = tid & 31, wid = tid >> 5;
  const int m    = lane & 15, half = lane >> 4;
  const int rg   = wid >> 2;            // row-group 0/1
  const int wq   = wid & 3;             // quarter within row-group
  const int i0   = blockIdx.x * 32;
  const int b    = blockIdx.y;
  const unsigned short* xb = x + (size_t)b * NN * DD;
  const unsigned short* qb = q + (size_t)b * NN * DD;
  const float* adjb        = adj + (size_t)b * NN * NN;

  if (tid < 32) rowsum[tid] = 0.f;
  if (wid == 0) {  // stage q: 32x256 via TDM (pad 16B per 512B row)
    tdm_load_2d(qb + (size_t)i0 * DD, lds_off(&q_lds[0]), DD, 32, DD, NN, DD, 6, 3);
    __builtin_amdgcn_s_wait_tensorcnt(0);
  }
  __syncthreads();

  v8f o[4] = {};
  for (int jb = 0; jb < NN; jb += 64) {
    if (wid == 0) {  // stage x rows jb..jb+63 via TDM
      tdm_load_2d(xb + (size_t)jb * DD, lds_off(&x_lds[0]), DD, 64, DD, NN, DD, 6, 3);
      __builtin_amdgcn_s_wait_tensorcnt(0);
    }
    __syncthreads();

    // phase 1: S(16x16) = q(rows rg) . x(rows jb+wq*16..)^T over K=256
    v8f s = {};
    #pragma unroll
    for (int kk = 0; kk < 8; ++kk) {
      int kbase = kk * 32;
      Frag a, bf_;
      int ar = rg * 16 + m;
      a.h[0]   = *(const us8*)&q_lds[ar * QS + kbase + half * 8];
      a.h[1]   = *(const us8*)&q_lds[ar * QS + kbase + 16 + half * 8];
      int jr = wq * 16 + m;        // B[k=d][n=jcol] = x[jcol][d]
      bf_.h[0] = *(const us8*)&x_lds[jr * XS + kbase + half * 8];
      bf_.h[1] = *(const us8*)&x_lds[jr * XS + kbase + 16 + half * 8];
      s = __builtin_amdgcn_wmma_f32_16x16x32_bf16(
              false, a.bf, false, bf_.bf, (short)0, s, false, false);
    }
    // sigmoid + diag eps + adj mask + rowsum + store S bf16
    #pragma unroll
    for (int r = 0; r < 8; ++r) {
      int ri = rg * 16 + r + 8 * half;
      int ig = i0 + ri;
      int jg = jb + wq * 16 + m;
      float a = 1.0f / (1.0f + __expf(-s[r]));
      bool diag = (ig == jg);
      if (diag) a += 1e-5f;
      a *= diag ? 1.0f : adjb[(size_t)ig * NN + jg];
      atomicAdd(&rowsum[ri], a);
      s_lds[ri * SS + wq * 16 + m] = f2bf(a);
    }
    __syncthreads();

    // phase 2: O(16 x 64) += S(16x64) @ x(64 x dcols), dcols = wq*64..
    #pragma unroll
    for (int kk = 0; kk < 2; ++kk) {
      int kbase = kk * 32;
      Frag a;
      int ar = rg * 16 + m;
      a.h[0] = *(const us8*)&s_lds[ar * SS + kbase + half * 8];
      a.h[1] = *(const us8*)&s_lds[ar * SS + kbase + 16 + half * 8];
      #pragma unroll
      for (int t = 0; t < 4; ++t) {
        Frag bb;  // B[k=jrow][n=dcol] = x_lds[k][d] (K varies per element)
        int d = wq * 64 + t * 16 + m;
        #pragma unroll
        for (int e = 0; e < 16; ++e) {
          int ke = (e < 8) ? (half * 8 + e) : (16 + half * 8 + (e - 8));
          bb.s[e] = x_lds[(kbase + ke) * XS + d];
        }
        o[t] = __builtin_amdgcn_wmma_f32_16x16x32_bf16(
                   false, a.bf, false, bb.bf, (short)0, o[t], false, false);
      }
    }
    __syncthreads();
  }

  // normalize rows and store (bf16)
  unsigned short* ob = out + (size_t)b * NN * DD;
  #pragma unroll
  for (int r = 0; r < 8; ++r) {
    int ri = rg * 16 + r + 8 * half;
    int rgl = i0 + ri;
    float inv = 1.0f / rowsum[ri];
    #pragma unroll
    for (int t = 0; t < 4; ++t) {
      int d = wq * 64 + t * 16 + m;
      ob[(size_t)rgl * DD + d] = f2bf(o[t][r] * inv);
    }
  }
}

// ---------------------------------------------------------------------------
extern "C" void kernel_launch(void* const* d_in, const int* in_sizes, int n_in,
                              void* d_out, int out_size, void* d_ws, size_t ws_size,
                              hipStream_t stream)
{
  const float* x   = (const float*)d_in[0];
  const float* adj = (const float*)d_in[1];
  const float* waw = (const float*)d_in[2];
  const float* wab = (const float*)d_in[3];
  const float* l0w = (const float*)d_in[4];
  const float* l0b = (const float*)d_in[5];
  const float* l1w = (const float*)d_in[6];
  const float* l1b = (const float*)d_in[7];
  const float* wf  = (const float*)d_in[8];
  const float* bf  = (const float*)d_in[9];
  float* out = (float*)d_out;

  const size_t xe = (size_t)BATCH * NN * DD;       // 8,388,608
  const size_t we = (size_t)LL * DD * DD;          // 262,144

  float* F0 = (float*)d_ws;                        // f32 x (residual chain)
  float* F1 = F0 + xe;
  unsigned short* B0   = (unsigned short*)(F1 + xe);   // bf16 x
  unsigned short* B1   = B0 + xe;                      // bf16 scratch (q / h)
  unsigned short* B2   = B1 + xe;                      // bf16 scratch (agg / x')
  unsigned short* wb_a = B2 + xe;                      // bf16 weights
  unsigned short* wb_0 = wb_a + we;
  unsigned short* wb_1 = wb_0 + we;
  unsigned short* wb_f = wb_1 + we;

  (void)hipMemcpyAsync(F0, x, xe * sizeof(float), hipMemcpyDeviceToDevice, stream);

  // one-time bf16 conversions
  f32_to_bf16<<<dim3(xe / 1024),       dim3(256), 0, stream>>>(x,   B0);
  f32_to_bf16<<<dim3(we / 1024),       dim3(256), 0, stream>>>(waw, wb_a);
  f32_to_bf16<<<dim3(we / 1024),       dim3(256), 0, stream>>>(l0w, wb_0);
  f32_to_bf16<<<dim3(we / 1024),       dim3(256), 0, stream>>>(l1w, wb_1);
  f32_to_bf16<<<dim3(DD * DD / 1024),  dim3(256), 0, stream>>>(wf,  wb_f);

  dim3 gblk(256), ggrid(MTOT / 64, DD / 128);
  dim3 ablk(256), agrid(NN / 32, BATCH);

  float* curf = F0; float* altf = F1;
  unsigned short *curb = B0, *t1 = B1, *t2 = B2;

  for (int l = 0; l < LL; ++l) {
    // q = x @ Wa^T + ba            (bf16 only: consumed by attention)
    gemm_bf16<<<ggrid, gblk, 0, stream>>>(curb, wb_a + (size_t)l * DD * DD,
                                          wab + (size_t)l * DD, nullptr,
                                          nullptr, t1, 0);
    // agg = rownorm((sigmoid(q x^T)+eps I)*adj) @ x   (bf16 only)
    gat_attn_agg<<<agrid, ablk, 0, stream>>>(t1, curb, adj, t2);
    // h = relu(agg @ W0^T + b0)    (bf16 only)
    gemm_bf16<<<ggrid, gblk, 0, stream>>>(t2, wb_0 + (size_t)l * DD * DD,
                                          l0b + (size_t)l * DD, nullptr,
                                          nullptr, t1, 1);
    // x' = relu(h @ W1^T + b1) + x   (f32 for residual chain + bf16 for matmuls)
    gemm_bf16<<<ggrid, gblk, 0, stream>>>(t1, wb_1 + (size_t)l * DD * DD,
                                          l1b + (size_t)l * DD, curf,
                                          altf, t2, 1);
    { float* tf = curf; curf = altf; altf = tf; }
    { unsigned short* tb = curb; curb = t2; t2 = tb; }
  }
  // out = x @ wf^T + bf  (f32 to d_out)
  gemm_bf16<<<ggrid, gblk, 0, stream>>>(curb, wb_f, bf, nullptr, out, nullptr, 0);
}
